// KPConvLayer_71116068488052
// MI455X (gfx1250) — matrix operand order repaired and checked by reading, not compiled, not run on hardware
//
#include <hip/hip_runtime.h>

typedef __attribute__((ext_vector_type(16))) _Float16 v16h;
typedef __attribute__((ext_vector_type(4)))  _Float16 v4h;
typedef __attribute__((ext_vector_type(8)))  float    v8f;

#define SPTS   15      // kernel points
#define KNB    32      // neighbors
#define CIN    128
#define COUT   128
#define NPTS   32768
#define BATCH  2
#define TILE_P 16      // points per workgroup
#define KT2    60      // GEMM2 K-tiles = 15*128/32

// LDS layout (units: _Float16 halves)
// A1: kw^T A-fragments, 16 points * 512 halves            = 8192
// A2: agg A-fragments for GEMM2, 60 ktiles * 512 halves   = 30720
// NF: neighbor-feature B-fragments, 8 pts * 8 ct * 512    = 32768
#define A1_OFF 0
#define A2_OFF 8192
#define NF_OFF (8192 + 30720)
#define LDS_HALVES (8192 + 30720 + 32768)   // 71680 halves = 143360 bytes

// workspace layout (bytes)
#define WF_BYTES   491520u                  // 60*8*512 halves * 2B
#define FEATH_OFF  524288u                  // 512 KB aligned start of f16 features
#define FEATH_BYTES ((size_t)BATCH * NPTS * CIN * 2)

// ---------------------------------------------------------------------------
// Pack fp32 weights (S, C_in, C_out) into f16 WMMA B-fragments in workspace.
// B layout (16-bit, 32x16): lane = (o&15) + 16*(k32>>4), half = k32&15.
// ---------------------------------------------------------------------------
__global__ __launch_bounds__(256) void kpconv_pack_w(
    const float* __restrict__ wts, _Float16* __restrict__ wfrag)
{
    int idx  = blockIdx.x * 256 + threadIdx.x;   // 60*8*32*16 = 245760 total
    int h    = idx & 15;
    int lane = (idx >> 4) & 31;
    int ot   = (idx >> 9) & 7;
    int j    = idx >> 12;                        // 0..59
    int g    = lane >> 4;
    int k32  = 16 * g + h;
    int kk   = j * 32 + k32;                     // flattened K index (s*128 + c)
    int s    = kk >> 7;
    int c    = kk & 127;
    int o    = ot * 16 + (lane & 15);
    wfrag[idx] = (_Float16)wts[(s * CIN + c) * COUT + o];
}

// ---------------------------------------------------------------------------
// One-shot fp32 -> f16 conversion of the feature table (amortized ~32x by the
// neighbor gather; halves the dominant HBM term of the whole kernel).
// ---------------------------------------------------------------------------
__global__ __launch_bounds__(256) void kpconv_cvt_feat(
    const float* __restrict__ f, _Float16* __restrict__ o)
{
    int i = blockIdx.x * 256 + threadIdx.x;      // one float4 per thread
    float4 v = ((const float4*)f)[i];
    v4h hv = { (_Float16)v.x, (_Float16)v.y, (_Float16)v.z, (_Float16)v.w };
    *(v4h*)(o + (size_t)i * 4) = hv;
}

// ---------------------------------------------------------------------------
// Main fused kernel: one 16-point tile per workgroup (256 thr = 8 waves).
// ---------------------------------------------------------------------------
template <bool USE_H>
__global__ __launch_bounds__(256) void kpconv_main(
    const float* __restrict__ xyz,  const float* __restrict__ featF,
    const _Float16* __restrict__ featH,
    const int*   __restrict__ nidx, const float* __restrict__ kpts,
    const _Float16* __restrict__ wfrag, const float* __restrict__ bias,
    float* __restrict__ out)
{
    extern __shared__ _Float16 sm[];
    _Float16* A1 = sm + A1_OFF;
    _Float16* A2 = sm + A2_OFF;
    _Float16* NF = sm + NF_OFF;

    const int tid  = threadIdx.x;
    const int lane = tid & 31;
    const int w    = tid >> 5;                    // wave id 0..7
    const int tile = blockIdx.x;
    const int b    = tile / (NPTS / TILE_P);
    const int n0   = (tile % (NPTS / TILE_P)) * TILE_P;
    const long rowBase = (long)b * NPTS + n0;

    // ---- zero the pad row (s=15) of every kw A-fragment -------------------
    if (tid < 32) {
        int p  = tid >> 1;
        int lz = 15 + 16 * (tid & 1);
        #pragma unroll
        for (int h = 0; h < 16; ++h)
            A1[p * 512 + lz * 16 + h] = (_Float16)0.0f;
    }

    // ---- phase 1: kernel correlations kw -> A1 (A-frag: rows=s, cols=k) ---
    const float inv2s2 = 555.5555556f;            // 1/(2*0.03^2)
    for (int pair = tid; pair < TILE_P * KNB; pair += 256) {
        int p = pair >> 5, k = pair & 31;
        long row  = rowBase + p;
        int  nb   = nidx[row * KNB + k];
        long nrow = (long)b * NPTS + nb;
        float rx = xyz[nrow * 3 + 0] - xyz[row * 3 + 0];
        float ry = xyz[nrow * 3 + 1] - xyz[row * 3 + 1];
        float rz = xyz[nrow * 3 + 2] - xyz[row * 3 + 2];
        // A-frag (16-bit): lane = s + 16*(k>>4); half = k&15
        int lgrp = k >> 4;
        int h    = k & 15;
        #pragma unroll
        for (int s = 0; s < SPTS; ++s) {
            float dx = rx - kpts[s * 3 + 0];
            float dy = ry - kpts[s * 3 + 1];
            float dz = rz - kpts[s * 3 + 2];
            float d2 = dx * dx + dy * dy + dz * dz;
            float kv = __expf(-d2 * inv2s2);
            A1[p * 512 + (s + 16 * lgrp) * 16 + h] = (_Float16)kv;
        }
    }

    const v8f vzero = {0.f, 0.f, 0.f, 0.f, 0.f, 0.f, 0.f, 0.f};

    // ---- two halves of 8 points: gather features, then GEMM1 --------------
    for (int half = 0; half < 2; ++half) {
        {   // gather: one neighbor-feature row per thread, store as B-frags
            int pl = tid >> 5;                    // 0..7 local point
            int k  = tid & 31;
            int p  = half * 8 + pl;
            long row = rowBase + p;
            int  nb  = nidx[row * KNB + k];
            int lghi = 16 * (k >> 4);             // B-frag lane group from K
            int h    = k & 15;                    // B-frag half from K
            _Float16* dst = NF + pl * 4096;
            if constexpr (USE_H) {
                const v16h* srcv =
                    (const v16h*)(featH + ((long)b * NPTS + nb) * CIN);
                #pragma unroll
                for (int c16 = 0; c16 < 8; ++c16) {
                    v16h v = srcv[c16];
                    #pragma unroll
                    for (int i = 0; i < 16; ++i)
                        dst[c16 * 512 + (i + lghi) * 16 + h] = v[i];
                }
            } else {
                const float4* src =
                    (const float4*)(featF + ((long)b * NPTS + nb) * CIN);
                #pragma unroll
                for (int c4 = 0; c4 < 32; ++c4) {
                    float4 v = src[c4];
                    int c = c4 * 4;
                    dst[((c+0)>>4)*512 + (((c+0)&15)+lghi)*16 + h] = (_Float16)v.x;
                    dst[((c+1)>>4)*512 + (((c+1)&15)+lghi)*16 + h] = (_Float16)v.y;
                    dst[((c+2)>>4)*512 + (((c+2)&15)+lghi)*16 + h] = (_Float16)v.z;
                    dst[((c+3)>>4)*512 + (((c+3)&15)+lghi)*16 + h] = (_Float16)v.w;
                }
            }
        }
        __syncthreads();

        {   // GEMM1: wave w handles point p = half*8 + w
            int p = half * 8 + w;
            const v16h a = *(const v16h*)(A1 + p * 512 + lane * 16);
            v8f acc[8];
            #pragma unroll
            for (int t = 0; t < 8; ++t) {
                v16h bf = *(const v16h*)(NF + w * 4096 + t * 512 + lane * 16);
                acc[t] = __builtin_amdgcn_wmma_f32_16x16x32_f16(
                    false, a, false, bf, (short)0, vzero, false, false);
            }
            // scatter agg (f16) into GEMM2 A-fragments
            int g   = lane >> 4;
            int clo = lane & 15;
            #pragma unroll
            for (int t = 0; t < 8; ++t) {
                #pragma unroll
                for (int r = 0; r < 8; ++r) {
                    int s = r + 8 * g;            // C/D layout: M = r (+8 hi lanes)
                    if (s < SPTS) {
                        int kk  = s * CIN + (t * 16 + clo);
                        int j   = kk >> 5, k32 = kk & 31;
                        A2[j * 512 + (p + 16 * (k32 >> 4)) * 16
                                   + (k32 & 15)] = (_Float16)acc[t][r];
                    }
                }
            }
        }
        __syncthreads();
    }

    // ---- GEMM2: out(16x128) = aggFlat(16x1920) @ W(1920x128) --------------
    v8f acc2 = vzero;
    const v16h* wf = (const v16h*)wfrag;
    #pragma unroll 4
    for (int j = 0; j < KT2; ++j) {
        v16h a2 = *(const v16h*)(A2 + j * 512 + lane * 16);
        v16h bw = wf[(size_t)(j * 8 + w) * 32 + lane];   // 32B coalesced frag
        acc2 = __builtin_amdgcn_wmma_f32_16x16x32_f16(
            false, a2, false, bw, (short)0, acc2, false, false);
    }

    // ---- epilogue: bias + f32 store ---------------------------------------
    int olo = lane & 15, g = lane >> 4;
    int o   = w * 16 + olo;
    float bv = bias[o];
    #pragma unroll
    for (int r = 0; r < 8; ++r) {
        int pp = r + 8 * g;                       // C/D layout: M = point
        out[(rowBase + pp) * COUT + o] = acc2[r] + bv;
    }
}

// ---------------------------------------------------------------------------
extern "C" void kernel_launch(void* const* d_in, const int* in_sizes, int n_in,
                              void* d_out, int out_size, void* d_ws, size_t ws_size,
                              hipStream_t stream)
{
    (void)in_sizes; (void)n_in; (void)out_size;
    const float* xyz  = (const float*)d_in[0];
    const float* feat = (const float*)d_in[1];
    const int*   nidx = (const int*)d_in[2];
    const float* kpts = (const float*)d_in[3];
    const float* wts  = (const float*)d_in[4];
    const float* bias = (const float*)d_in[5];
    float* out = (float*)d_out;

    _Float16* wfrag = (_Float16*)d_ws;            // 480 KB of fragments
    _Float16* featH = (_Float16*)((char*)d_ws + FEATH_OFF);
    const bool useH = ws_size >= (size_t)FEATH_OFF + FEATH_BYTES;

    // pack weights into WMMA B-fragments (weights then stay hot in L2)
    kpconv_pack_w<<<960, 256, 0, stream>>>(wts, wfrag);

    const int tiles = (BATCH * NPTS) / TILE_P;    // 4096 workgroups
    const size_t lds_bytes = LDS_HALVES * sizeof(_Float16);   // 143360 B

    if (useH) {
        // fp32 -> f16 feature table: halves the dominant random-gather HBM term
        const int n4 = (BATCH * NPTS * CIN) / 4;  // 2,097,152 float4s
        kpconv_cvt_feat<<<n4 / 256, 256, 0, stream>>>(feat, featH);
        kpconv_main<true><<<tiles, 256, lds_bytes, stream>>>(
            xyz, feat, featH, nidx, kpts, wfrag, bias, out);
    } else {
        kpconv_main<false><<<tiles, 256, lds_bytes, stream>>>(
            xyz, feat, featH, nidx, kpts, wfrag, bias, out);
    }
}
